// SRWINBlock_27032524161142
// MI455X (gfx1250) — matrix operand order repaired and checked by reading, not compiled, not run on hardware
//
#include <hip/hip_runtime.h>
#include <math.h>

// ---------------------------------------------------------------------------
// CDNA5 (gfx1250) implementation of the SRWin block.
// All GEMMs + attention einsums use v_wmma_f32_16x16x32_bf16 (wave32 WMMA).
// bf16 GEMM A-tiles are staged into LDS with the Tensor Data Mover
// (tensor_load_to_lds + s_wait_tensorcnt) when the builtin is available.
// ---------------------------------------------------------------------------

typedef __bf16 bf16_t;
typedef __attribute__((ext_vector_type(16))) __bf16 v16bf;
typedef __attribute__((ext_vector_type(8)))  float  v8f;
typedef unsigned int u32x4 __attribute__((ext_vector_type(4)));
typedef int          i32x8 __attribute__((ext_vector_type(8)));
typedef int          i32x4 __attribute__((ext_vector_type(4)));

#define BATCH   4
#define HDIM    256
#define WDIM    256
#define LTOT    (HDIM * WDIM)          // 65536
#define BL      (BATCH * LTOT)         // 262144
#define CDIM    192
#define LBIAS   465                    // (2*8-1)*(2*16-1)

#if __has_builtin(__builtin_amdgcn_tensor_load_to_lds) && \
    __has_builtin(__builtin_amdgcn_s_wait_tensorcnt)
#define HAVE_TDM 1
#else
#define HAVE_TDM 0
#endif

#if HAVE_TDM
// ---------------------------------------------------------------------------
// Tensor Data Mover: 2D tile load (bf16, raw copy) global -> LDS.
// D# packing per CDNA5 ISA 8.3/8.4 (normal mode, 2D, no pad/iterate/multicast).
// ---------------------------------------------------------------------------
__device__ __forceinline__ void tdm_load_tile_bf16(
    unsigned int lds_off, const bf16_t* gsrc,
    unsigned int tensor_d0, unsigned int tensor_d1,
    unsigned int tile_d0, unsigned int tile_d1, unsigned int stride_elems) {
  const unsigned long long ga = (unsigned long long)(uintptr_t)gsrc;
  u32x4 g0;
  g0[0] = 1u;                                              // count=1, user D#
  g0[1] = lds_off;                                         // lds_addr (bytes)
  g0[2] = (unsigned int)(ga & 0xffffffffu);                // global_addr[31:0]
  g0[3] = (unsigned int)((ga >> 32) & 0x01ffffffu)         // global_addr[56:32]
          | (2u << 30);                                    // type = 2 ("image")
  i32x8 g1;
  g1[0] = (int)(1u << 16);                                 // data_size=1 (2B), mask=0
  g1[1] = (int)((tensor_d0 & 0xffffu) << 16);              // tensor_dim0[15:0] @63:48
  g1[2] = (int)((tensor_d0 >> 16) |                        // tensor_dim0[31:16]
                ((tensor_d1 & 0xffffu) << 16));            // tensor_dim1[15:0]
  g1[3] = (int)((tensor_d1 >> 16) |                        // tensor_dim1[31:16]
                ((tile_d0 & 0xffffu) << 16));              // tile_dim0
  g1[4] = (int)(tile_d1 & 0xffffu);                        // tile_dim1; tile_dim2=0 (2D)
  g1[5] = (int)stride_elems;                               // tensor_dim0_stride[31:0]
  g1[6] = 0;                                               // stride hi / dim1_stride lo
  g1[7] = 0;                                               // dim1_stride hi
  const i32x4 z4 = {0, 0, 0, 0};
#if defined(__clang_major__) && (__clang_major__ >= 23)
  const i32x8 z8 = {0, 0, 0, 0, 0, 0, 0, 0};
  __builtin_amdgcn_tensor_load_to_lds(g0, g1, z4, z4, z8, 0);
#else
  __builtin_amdgcn_tensor_load_to_lds(g0, g1, z4, z4, 0);
#endif
}
#endif  // HAVE_TDM

// ---------------------------------------------------------------------------
// WMMA fragment loaders (wave32; A/B use this file's contiguous k order, not the ISA table order)
// A 16x32 bf16: lane m = l&15; halves e0..15 -> K = e + 16*(l>=16).
// B 32x16 bf16: lane n = l&15; halves e0..15 -> K = e + 16*(l>=16).
// D 16x16 f32 : lane n = l&15; vgpr r -> M = r + 8*(l>=16).
// ---------------------------------------------------------------------------
__device__ __forceinline__ v16bf load_afrag(const bf16_t* base, int ldk, int lane) {
  const int m  = lane & 15;
  const int kb = (lane >> 4) << 4;
  const bf16_t* p = base + m * ldk + kb;
  v16bf a;
#pragma unroll
  for (int i = 0; i < 16; ++i) a[i] = p[i];
  return a;
}

__device__ __forceinline__ v16bf load_bfrag(const bf16_t* base, int ldk, int lane) {
  const int n  = lane & 15;
  const int kb = (lane >> 4) << 4;
  const bf16_t* p = base + n * ldk + kb;
  v16bf b;
#pragma unroll
  for (int i = 0; i < 16; ++i) b[i] = p[i];
  return b;
}

__device__ __forceinline__ v8f wmma_bf16(v16bf a, v16bf b, v8f c) {
  return __builtin_amdgcn_wmma_f32_16x16x32_bf16(false, a, false, b, (short)0, c,
                                                 false, false);
}

__device__ __forceinline__ v8f vzero8() {
  v8f z = {0.f, 0.f, 0.f, 0.f, 0.f, 0.f, 0.f, 0.f};
  return z;
}

// ---------------------------------------------------------------------------
// Generic WMMA GEMM: out[M,N] = epi(A[M,K] @ W[K,N] + bias (+ res))
// Block tile 128x64, BK=32, 256 threads = 8 waves in 4(M) x 2(N), each wave
// computes a 32x32 patch (2x2 WMMA tiles).
// AF32: A is fp32 (converted to bf16 in LDS) else A is bf16 (TDM-staged).
// EPI : 0 -> store f32, 1 -> store bf16, 2 -> exact GELU then store bf16.
// RES : add res[M,N] (fp32) before epilogue store.
// ---------------------------------------------------------------------------
template <bool AF32, int EPI, bool RES>
__global__ __launch_bounds__(256)
void gemm_wmma_kernel(const void* __restrict__ Ain, const float* __restrict__ Wg,
                      const float* __restrict__ bias, const float* __restrict__ res,
                      void* __restrict__ outp, int M, int N, int K) {
  __shared__ bf16_t sA[128][32];   // [m][k]
  __shared__ bf16_t sBt[64][32];   // [n][k] (transposed W tile)

  const int tid  = threadIdx.x;
  const int lane = tid & 31;
  const int wave = tid >> 5;
  const int mB = blockIdx.x * 128;
  const int nB = blockIdx.y * 64;
  const int wm = (wave >> 1) << 5;   // 0,32,64,96
  const int wn = (wave & 1) << 5;    // 0,32

  v8f acc[2][2];
#pragma unroll
  for (int i = 0; i < 2; ++i)
#pragma unroll
    for (int j = 0; j < 2; ++j) acc[i][j] = vzero8();

  const int arow = tid >> 1;          // 0..127
  const int acol = (tid & 1) << 4;    // 0 / 16
  const int bk   = tid >> 3;          // 0..31
  const int bcol = (tid & 7) << 3;    // 0..56

#if HAVE_TDM
  const unsigned int ldsA_off = (unsigned int)(uintptr_t)(&sA[0][0]);
#endif

  for (int k0 = 0; k0 < K; k0 += 32) {
    if (AF32) {
      const float* A = (const float*)Ain;
      const float* src = A + (size_t)(mB + arow) * K + k0 + acol;
#pragma unroll
      for (int i = 0; i < 16; ++i) sA[arow][acol + i] = (bf16_t)src[i];
      if (k0 + 32 < K) __builtin_prefetch(src + 32, 0, 1);
    } else {
#if HAVE_TDM
      // Async-tensor copy of the whole 128x32 bf16 tile; single TDM op
      // issued by wave 0 (EXEC-ignored, wave-level instruction).
      if (wave == 0) {
        const bf16_t* A = (const bf16_t*)Ain;
        tdm_load_tile_bf16(ldsA_off, A + (size_t)mB * K + k0,
                           (unsigned int)K, (unsigned int)M,
                           32u, 128u, (unsigned int)K);
        __builtin_amdgcn_s_wait_tensorcnt(0);
      }
#else
      const bf16_t* A = (const bf16_t*)Ain;
      const bf16_t* src = A + (size_t)(mB + arow) * K + k0 + acol;
#pragma unroll
      for (int i = 0; i < 16; ++i) sA[arow][acol + i] = src[i];
      if (k0 + 32 < K) __builtin_prefetch(src + 32, 0, 1);
#endif
    }
    {
      const float* src = Wg + (size_t)(k0 + bk) * N + nB + bcol;
#pragma unroll
      for (int i = 0; i < 8; ++i) sBt[bcol + i][bk] = (bf16_t)src[i];
    }
    __syncthreads();

    const v16bf b0 = load_bfrag(&sBt[wn][0], 32, lane);
    const v16bf b1 = load_bfrag(&sBt[wn + 16][0], 32, lane);
    const v16bf a0 = load_afrag(&sA[wm][0], 32, lane);
    const v16bf a1 = load_afrag(&sA[wm + 16][0], 32, lane);
    acc[0][0] = wmma_bf16(a0, b0, acc[0][0]);
    acc[0][1] = wmma_bf16(a0, b1, acc[0][1]);
    acc[1][0] = wmma_bf16(a1, b0, acc[1][0]);
    acc[1][1] = wmma_bf16(a1, b1, acc[1][1]);
    __syncthreads();
  }

  const int hl  = lane >> 4;
  const int l15 = lane & 15;
#pragma unroll
  for (int jn = 0; jn < 2; ++jn) {
    const int ng = nB + wn + (jn << 4) + l15;
    const float bval = bias[ng];
#pragma unroll
    for (int im = 0; im < 2; ++im) {
#pragma unroll
      for (int r = 0; r < 8; ++r) {
        const int mg = mB + wm + (im << 4) + r + (hl << 3);
        float v = acc[im][jn][r] + bval;
        if (RES) v += res[(size_t)mg * N + ng];
        if (EPI == 2) v = 0.5f * v * (1.0f + erff(v * 0.70710678118654752f));
        if (EPI == 0) ((float*)outp)[(size_t)mg * N + ng] = v;
        else          ((bf16_t*)outp)[(size_t)mg * N + ng] = (bf16_t)v;
      }
    }
  }
}

// ---------------------------------------------------------------------------
// LayerNorm over C=192, one wave (32 lanes x 6 elems) per row -> bf16 out
// ---------------------------------------------------------------------------
__global__ __launch_bounds__(256)
void ln_kernel(const float* __restrict__ x, const float* __restrict__ g,
               const float* __restrict__ be, bf16_t* __restrict__ out) {
  const int row  = blockIdx.x * 8 + (threadIdx.x >> 5);
  const int lane = threadIdx.x & 31;
  const float* xr = x + (size_t)row * CDIM;
  float v[6];
  float s = 0.f;
#pragma unroll
  for (int i = 0; i < 6; ++i) { v[i] = xr[lane + i * 32]; s += v[i]; }
#pragma unroll
  for (int d = 1; d < 32; d <<= 1) s += __shfl_xor(s, d, 32);
  const float mu = s * (1.0f / 192.0f);
  float q = 0.f;
#pragma unroll
  for (int i = 0; i < 6; ++i) { const float dd = v[i] - mu; q += dd * dd; }
#pragma unroll
  for (int d = 1; d < 32; d <<= 1) q += __shfl_xor(q, d, 32);
  const float rs = rsqrtf(q * (1.0f / 192.0f) + 1e-5f);
  bf16_t* orow = out + (size_t)row * CDIM;
#pragma unroll
  for (int i = 0; i < 6; ++i) {
    const int c = lane + i * 32;
    orow[c] = (bf16_t)((v[i] - mu) * rs * g[c] + be[c]);
  }
}

// ---------------------------------------------------------------------------
// Dynamic positional bias MLP (2 -> 6 -> 6 -> 6 -> 4), both branches
// ---------------------------------------------------------------------------
struct PosParams {
  const float* p[14];  // proj_w, proj_b, ln1_g, ln1_b, w1, b1, ln2_g, ln2_b,
                       // w2, b2, ln3_g, ln3_b, w3, b3
};

__device__ __forceinline__ void ln6_relu(float* h, const float* g, const float* b) {
  float mu = 0.f;
#pragma unroll
  for (int j = 0; j < 6; ++j) mu += h[j];
  mu *= (1.0f / 6.0f);
  float var = 0.f;
#pragma unroll
  for (int j = 0; j < 6; ++j) { const float d = h[j] - mu; var += d * d; }
  var *= (1.0f / 6.0f);
  const float rs = rsqrtf(var + 1e-5f);
#pragma unroll
  for (int j = 0; j < 6; ++j) h[j] = fmaxf((h[j] - mu) * rs * g[j] + b[j], 0.f);
}

__global__ void pos_table_kernel(const float* __restrict__ biases_h,
                                 const float* __restrict__ biases_v,
                                 PosParams ph, PosParams pv,
                                 float* __restrict__ table) {
  const int r = blockIdx.x * 256 + threadIdx.x;
  if (r >= 2 * LBIAS) return;
  const int branch = (r >= LBIAS) ? 1 : 0;
  const int li = branch ? (r - LBIAS) : r;
  const float* bi = (branch ? biases_v : biases_h) + li * 2;
  const PosParams P = branch ? pv : ph;
  const float* proj_w = P.p[0];  const float* proj_b = P.p[1];
  const float* ln1_g  = P.p[2];  const float* ln1_b  = P.p[3];
  const float* w1     = P.p[4];  const float* b1     = P.p[5];
  const float* ln2_g  = P.p[6];  const float* ln2_b  = P.p[7];
  const float* w2     = P.p[8];  const float* b2     = P.p[9];
  const float* ln3_g  = P.p[10]; const float* ln3_b  = P.p[11];
  const float* w3     = P.p[12]; const float* b3     = P.p[13];

  float h[6], t[6];
#pragma unroll
  for (int j = 0; j < 6; ++j) h[j] = bi[0] * proj_w[j] + bi[1] * proj_w[6 + j] + proj_b[j];
  ln6_relu(h, ln1_g, ln1_b);
#pragma unroll
  for (int j = 0; j < 6; ++j) {
    float a = b1[j];
#pragma unroll
    for (int i = 0; i < 6; ++i) a += h[i] * w1[i * 6 + j];
    t[j] = a;
  }
#pragma unroll
  for (int j = 0; j < 6; ++j) h[j] = t[j];
  ln6_relu(h, ln2_g, ln2_b);
#pragma unroll
  for (int j = 0; j < 6; ++j) {
    float a = b2[j];
#pragma unroll
    for (int i = 0; i < 6; ++i) a += h[i] * w2[i * 6 + j];
    t[j] = a;
  }
#pragma unroll
  for (int j = 0; j < 6; ++j) h[j] = t[j];
  ln6_relu(h, ln3_g, ln3_b);
#pragma unroll
  for (int j = 0; j < 4; ++j) {
    float a = b3[j];
#pragma unroll
    for (int i = 0; i < 6; ++i) a += h[i] * w3[i * 4 + j];
    table[((size_t)branch * LBIAS + li) * 4 + j] = a;
  }
}

// Expand table via relative-position index -> bias[branch][head][128][128]
__global__ void bias_expand_kernel(const int* __restrict__ rpi_h,
                                   const int* __restrict__ rpi_v,
                                   const float* __restrict__ table,
                                   float* __restrict__ bexp) {
  const int i = blockIdx.x * 256 + threadIdx.x;   // 0 .. 2*16384-1
  const int branch = i >> 14;
  const int idx = i & 16383;
  const int* rpi = branch ? rpi_v : rpi_h;
  const int r = rpi[idx];
  const float* tb = table + ((size_t)branch * LBIAS + r) * 4;
#pragma unroll
  for (int h = 0; h < 4; ++h)
    bexp[(((size_t)(branch * 4 + h)) << 14) + idx] = tb[h];
}

// ---------------------------------------------------------------------------
// Windowed attention: one block per (branch, window). 4 heads, hd 24 -> pad 32.
// S = scale*(Q K^T) + bias ; P = softmax(S) ; O = P V   (all via WMMA bf16)
// ---------------------------------------------------------------------------
__global__ __launch_bounds__(256)
void attn_kernel(const bf16_t* __restrict__ qkv, const float* __restrict__ bexp,
                 float* __restrict__ att) {
  __shared__ bf16_t lq[128][32];
  __shared__ bf16_t lk[128][32];
  __shared__ bf16_t lvT[32][128];   // [channel][token]
  __shared__ bf16_t lp[128][128];

  const int tid  = threadIdx.x;
  const int lane = tid & 31;
  const int wave = tid >> 5;       // wave = M-tile (query rows 16w..16w+15)
  const int branch = blockIdx.x >> 11;
  const int win = blockIdx.x & 2047;
  const int Hs = branch ? 16 : 8;
  const int Ws = branch ? 8 : 16;
  const int nWw = 256 / Ws;
  const int bidx = win >> 9;             // 512 windows per batch image
  const int wrem = win & 511;
  const int hb = wrem / nWw;
  const int wb = wrem - hb * nWw;
  const float scale = 0.20412414523193154f;  // 24^-0.5
  const int hl  = lane >> 4;
  const int l15 = lane & 15;

  const int lt  = tid >> 1;              // token loaded by this thread
  const int lc0 = (tid & 1) << 4;        // channel chunk 0 / 16
  const int lhs = lt / Ws, lws = lt - lhs * Ws;
  const size_t lrow = (size_t)bidx * LTOT + (size_t)(hb * Hs + lhs) * WDIM + (wb * Ws + lws);

  for (int h = 0; h < 4; ++h) {
    const int co = branch * 96 + h * 24;
    const bf16_t* gq = qkv + lrow * 576 + co;
#pragma unroll
    for (int i = 0; i < 16; ++i) {
      const int ch = lc0 + i;
      const bool ok = ch < 24;
      const bf16_t zq = ok ? gq[ch]       : (bf16_t)0.0f;
      const bf16_t zk = ok ? gq[192 + ch] : (bf16_t)0.0f;
      const bf16_t zv = ok ? gq[384 + ch] : (bf16_t)0.0f;
      lq[lt][ch] = zq;
      lk[lt][ch] = zk;
      lvT[ch][lt] = zv;
    }
    __syncthreads();

    // --- scores: wave owns rows [16w, 16w+16) x all 128 keys (8 tiles) ---
    v8f s[8];
#pragma unroll
    for (int nt = 0; nt < 8; ++nt) s[nt] = vzero8();
    const v16bf aq = load_afrag(&lq[wave * 16][0], 32, lane);
#pragma unroll
    for (int nt = 0; nt < 8; ++nt) {
      const v16bf bk = load_bfrag(&lk[nt * 16][0], 32, lane);
      s[nt] = wmma_bf16(aq, bk, s[nt]);
    }

    const float* bb = bexp + (((size_t)(branch * 4 + h)) << 14);
#pragma unroll
    for (int nt = 0; nt < 8; ++nt)
#pragma unroll
      for (int r = 0; r < 8; ++r) {
        const int m = wave * 16 + r + (hl << 3);
        const int n = nt * 16 + l15;
        s[nt][r] = s[nt][r] * scale + bb[m * 128 + n];
      }

    // --- softmax over 128 keys: 8 tiles + shuffle across 16-lane half ---
    float inv[8];
#pragma unroll
    for (int r = 0; r < 8; ++r) {
      float mx = -3.0e38f;
#pragma unroll
      for (int nt = 0; nt < 8; ++nt) mx = fmaxf(mx, s[nt][r]);
#pragma unroll
      for (int d = 1; d < 16; d <<= 1) mx = fmaxf(mx, __shfl_xor(mx, d, 32));
      float t = 0.f;
#pragma unroll
      for (int nt = 0; nt < 8; ++nt) {
        const float e = __expf(s[nt][r] - mx);
        s[nt][r] = e;
        t += e;
      }
#pragma unroll
      for (int d = 1; d < 16; d <<= 1) t += __shfl_xor(t, d, 32);
      inv[r] = 1.0f / t;
    }
#pragma unroll
    for (int nt = 0; nt < 8; ++nt)
#pragma unroll
      for (int r = 0; r < 8; ++r)
        lp[wave * 16 + r + (hl << 3)][nt * 16 + l15] = (bf16_t)(s[nt][r] * inv[r]);
    __syncthreads();

    // --- O = P V : K loop over 128 in 4 steps of 32, N = 32 (24 real) ---
    v8f o0 = vzero8(), o1 = vzero8();
#pragma unroll
    for (int ks = 0; ks < 4; ++ks) {
      const v16bf ap  = load_afrag(&lp[wave * 16][ks * 32], 128, lane);
      const v16bf bv0 = load_bfrag(&lvT[0][ks * 32], 128, lane);
      const v16bf bv1 = load_bfrag(&lvT[16][ks * 32], 128, lane);
      o0 = wmma_bf16(ap, bv0, o0);
      o1 = wmma_bf16(ap, bv1, o1);
    }
#pragma unroll
    for (int r = 0; r < 8; ++r) {
      const int m = wave * 16 + r + (hl << 3);   // token
      const int ohs = m / Ws, ows = m - ohs * Ws;
      const size_t orow = (size_t)bidx * LTOT + (size_t)(hb * Hs + ohs) * WDIM + (wb * Ws + ows);
      float* dst = att + orow * CDIM + co;
      dst[l15] = o0[r];
      if (l15 < 8) dst[16 + l15] = o1[r];
    }
    __syncthreads();
  }
}

// ---------------------------------------------------------------------------
// Depthwise 3x3 conv on V (read back from bf16 qkv) + bias, accumulated into att
// ---------------------------------------------------------------------------
__global__ __launch_bounds__(256)
void dwconv_kernel(const bf16_t* __restrict__ qkv, const float* __restrict__ wgt,
                   const float* __restrict__ bias, float* __restrict__ att) {
  const size_t i = (size_t)blockIdx.x * 256 + threadIdx.x;  // over BL*C
  const int c = (int)(i % CDIM);
  const size_t p = i / CDIM;
  const int w  = (int)(p % WDIM);
  const int hh = (int)((p / WDIM) % HDIM);
  const int bb = (int)(p / LTOT);
  float s = bias[c];
#pragma unroll
  for (int ky = 0; ky < 3; ++ky) {
    const int y = hh + ky - 1;
    if (y < 0 || y >= HDIM) continue;
#pragma unroll
    for (int kx = 0; kx < 3; ++kx) {
      const int x = w + kx - 1;
      if (x < 0 || x >= WDIM) continue;
      const size_t row = (size_t)bb * LTOT + (size_t)y * WDIM + x;
      s += (float)qkv[row * 576 + 384 + c] * wgt[c * 9 + ky * 3 + kx];
    }
  }
  att[i] += s;
}

// ---------------------------------------------------------------------------
// Host launcher
// ---------------------------------------------------------------------------
extern "C" void kernel_launch(void* const* d_in, const int* in_sizes, int n_in,
                              void* d_out, int out_size, void* d_ws, size_t ws_size,
                              hipStream_t stream) {
  (void)in_sizes; (void)out_size; (void)ws_size;
  const float* x        = (const float*)d_in[0];
  const int*   rpi_h    = (const int*)d_in[3];
  const int*   rpi_v    = (const int*)d_in[4];
  const float* biases_h = (const float*)d_in[5];
  const float* biases_v = (const float*)d_in[6];
  const float* norm1_g  = (const float*)d_in[7];
  const float* norm1_b  = (const float*)d_in[8];
  const float* qkv_w    = (const float*)d_in[9];
  const float* qkv_b    = (const float*)d_in[10];
  const float* getv_w   = (const float*)d_in[11];
  const float* getv_b   = (const float*)d_in[12];

  PosParams ph, pv;
  const float *proj_w, *proj_b, *norm2_g, *norm2_b, *fc1_w, *fc1_b, *fc2_w, *fc2_b;

  if (n_in >= 49) {
    // pos dicts flattened into 14 leaves each, sorted key order:
    // b1,b2,b3,ln1_b,ln1_g,ln2_b,ln2_g,ln3_b,ln3_g,proj_b,proj_w,w1,w2,w3
    auto fill = [&](PosParams& P, int base) {
      const float* L[14];
      for (int i = 0; i < 14; ++i) L[i] = (const float*)d_in[base + i];
      P.p[0]  = L[10]; P.p[1]  = L[9];  // proj_w, proj_b
      P.p[2]  = L[4];  P.p[3]  = L[3];  // ln1_g, ln1_b
      P.p[4]  = L[11]; P.p[5]  = L[0];  // w1, b1
      P.p[6]  = L[6];  P.p[7]  = L[5];  // ln2_g, ln2_b
      P.p[8]  = L[12]; P.p[9]  = L[1];  // w2, b2
      P.p[10] = L[8];  P.p[11] = L[7];  // ln3_g, ln3_b
      P.p[12] = L[13]; P.p[13] = L[2];  // w3, b3
    };
    fill(ph, 13);
    fill(pv, 27);
    proj_w = (const float*)d_in[41]; proj_b = (const float*)d_in[42];
    norm2_g = (const float*)d_in[43]; norm2_b = (const float*)d_in[44];
    fc1_w = (const float*)d_in[45]; fc1_b = (const float*)d_in[46];
    fc2_w = (const float*)d_in[47]; fc2_b = (const float*)d_in[48];
  } else {
    // pos dicts passed as one flat buffer each, insertion order
    auto fill = [&](PosParams& P, const float* b) {
      P.p[0] = b + 0;   P.p[1] = b + 12;    // proj_w(2x6), proj_b(6)
      P.p[2] = b + 18;  P.p[3] = b + 24;    // ln1_g, ln1_b
      P.p[4] = b + 30;  P.p[5] = b + 66;    // w1(6x6), b1
      P.p[6] = b + 72;  P.p[7] = b + 78;    // ln2_g, ln2_b
      P.p[8] = b + 84;  P.p[9] = b + 120;   // w2, b2
      P.p[10] = b + 126; P.p[11] = b + 132; // ln3_g, ln3_b
      P.p[12] = b + 138; P.p[13] = b + 162; // w3(6x4), b3(4)
    };
    fill(ph, (const float*)d_in[13]);
    fill(pv, (const float*)d_in[14]);
    proj_w = (const float*)d_in[15]; proj_b = (const float*)d_in[16];
    norm2_g = (const float*)d_in[17]; norm2_b = (const float*)d_in[18];
    fc1_w = (const float*)d_in[19]; fc1_b = (const float*)d_in[20];
    fc2_w = (const float*)d_in[21]; fc2_b = (const float*)d_in[22];
  }

  // Workspace layout (bytes):
  char* ws = (char*)d_ws;
  bf16_t* xn    = (bf16_t*)(ws);                       // BL*192  bf16  (100.7 MB)
  bf16_t* qkv   = (bf16_t*)(ws + 100663296);           // BL*576  bf16  (302.0 MB)
  float*  att   = (float*)(ws + 402653184);            // BL*192  f32   (201.3 MB)
  float*  table = (float*)(ws + 603979776);            // 2*465*4 f32
  float*  bexp  = (float*)(ws + 603996160);            // 2*4*128*128 f32
  bf16_t* hbuf  = qkv;                                 // reuse: BL*384 bf16
  bf16_t* xn2   = xn;                                  // reuse
  float*  x1    = (float*)d_out;

  // 1) dynamic pos-bias tables + expansion
  pos_table_kernel<<<4, 256, 0, stream>>>(biases_h, biases_v, ph, pv, table);
  bias_expand_kernel<<<128, 256, 0, stream>>>(rpi_h, rpi_v, table, bexp);
  // 2) LN1 -> bf16
  ln_kernel<<<BL / 8, 256, 0, stream>>>(x, norm1_g, norm1_b, xn);
  // 3) QKV GEMM (bf16 out)
  gemm_wmma_kernel<false, 1, false><<<dim3(BL / 128, 576 / 64), 256, 0, stream>>>(
      xn, qkv_w, qkv_b, nullptr, qkv, BL, 576, CDIM);
  // 4) windowed attention (both branches)
  attn_kernel<<<4096, 256, 0, stream>>>(qkv, bexp, att);
  // 5) depthwise 3x3 on V, accumulated into att
  dwconv_kernel<<<(BL * CDIM) / 256, 256, 0, stream>>>(qkv, getv_w, getv_b, att);
  // 6) proj GEMM + x residual -> x1 (written to d_out, fp32)
  gemm_wmma_kernel<true, 0, true><<<dim3(BL / 128, CDIM / 64), 256, 0, stream>>>(
      att, proj_w, proj_b, x, x1, BL, CDIM, CDIM);
  // 7) LN2 -> bf16
  ln_kernel<<<BL / 8, 256, 0, stream>>>(x1, norm2_g, norm2_b, xn2);
  // 8) FC1 + exact GELU -> bf16
  gemm_wmma_kernel<false, 2, false><<<dim3(BL / 128, 384 / 64), 256, 0, stream>>>(
      xn2, fc1_w, fc1_b, nullptr, hbuf, BL, 384, CDIM);
  // 9) FC2 + x1 residual -> d_out (fp32)
  gemm_wmma_kernel<false, 0, true><<<dim3(BL / 128, CDIM / 64), 256, 0, stream>>>(
      hbuf, fc2_w, fc2_b, x1, x1, BL, CDIM, 384);
}
